// Detector_1864015807078
// MI455X (gfx1250) — matrix operand order repaired and checked by reading, not compiled, not run on hardware
//
#include <hip/hip_runtime.h>
#include <hip/hip_bf16.h>

#define HD 128   // feature/hidden dim
#define NLAYERS 3

typedef __bf16 bf16_t;
typedef bf16_t v16bf __attribute__((ext_vector_type(16)));
typedef float  v8f   __attribute__((ext_vector_type(8)));

// ---------- helpers ----------
__device__ __forceinline__ unsigned f2bf_bits(float f) {
  unsigned u = __float_as_uint(f);
  return (u + 0x7FFFu + ((u >> 16) & 1u)) >> 16;   // round-to-nearest-even
}

// pack fp32 weight [128 x 128] (row-major, K x N) into bf16 K-pairs:
// Wp[(k/2)*128 + n] = { bf16(W[k][n]) , bf16(W[k+1][n]) }  (lo,hi)
// -> exactly the CDNA5 16-bit B-matrix VGPR pairing, one dword load per fragment reg.
__global__ void pack_weight_bf16(const float* __restrict__ W,
                                 unsigned* __restrict__ Wp, int total) {
  int idx = blockIdx.x * blockDim.x + threadIdx.x;
  if (idx >= total) return;
  int kh = idx >> 7;       // k/2
  int n  = idx & 127;
  unsigned lo = f2bf_bits(W[(2 * kh)     * HD + n]);
  unsigned hi = f2bf_bits(W[(2 * kh + 1) * HD + n]);
  Wp[idx] = lo | (hi << 16);
}

// ---------- degree / norm ----------
__global__ void deg_init(float* deg, int n) {
  int i = blockIdx.x * blockDim.x + threadIdx.x;
  if (i < n) deg[i] = 1.0f;                       // self-loop
}
__global__ void deg_accum(const int* __restrict__ dst, float* deg, int E) {
  int e = blockIdx.x * blockDim.x + threadIdx.x;
  if (e < E) atomicAdd(&deg[dst[e]], 1.0f);
}
__global__ void deg_rsqrt(float* deg, int n) {
  int i = blockIdx.x * blockDim.x + threadIdx.x;
  if (i < n) deg[i] = rsqrtf(deg[i]);             // deg >= 1 always
}
__global__ void edge_norm(const int* __restrict__ src, const int* __restrict__ dst,
                          const float* __restrict__ dinv, float* __restrict__ nrm, int E) {
  int e = blockIdx.x * blockDim.x + threadIdx.x;
  if (e < E) nrm[e] = dinv[src[e]] * dinv[dst[e]];
}

// ---------- GEMM: out[M x 128] = act(A[M x 128] @ W[128 x 128] + bias) ----------
// block = 256 threads = 8 waves; block owns 16 rows, wave w owns columns [16w,16w+16).
// A tile staged to LDS as bf16 pairs, row stride 67 dwords (bank-conflict free).
__global__ __launch_bounds__(256)
void gemm_bf16_wmma(const float* __restrict__ A,
                    const unsigned* __restrict__ Wp,   // packed bf16 pairs, 64*128 dwords
                    const float* __restrict__ bias,    // [128] or nullptr
                    float* __restrict__ out,
                    int do_relu) {
  __shared__ unsigned AsU[16 * 67];
  const int row0 = blockIdx.x * 16;

  // stage 16x128 fp32 -> bf16-pair dwords (1024 dwords, 4 per thread)
  for (int d = threadIdx.x; d < 16 * 64; d += 256) {
    int r  = d >> 6;
    int cp = d & 63;
    const float* ap = A + (size_t)(row0 + r) * HD + 2 * cp;
    AsU[r * 67 + cp] = f2bf_bits(ap[0]) | (f2bf_bits(ap[1]) << 16);
  }
  __syncthreads();

  const int lane = threadIdx.x & 31;
  const int wid  = threadIdx.x >> 5;
  const int n0   = wid * 16;
  const int m    = lane & 15;
  const int half = lane >> 4;

  v8f acc = {};
#pragma unroll
  for (int k0 = 0; k0 < HD; k0 += 32) {
    const int k0h = k0 >> 1;
    union { unsigned u[8]; v16bf v; } a, b;
    // A 16x32 bf16 layout: VGPR g <-> K = (g>=4?16:0) + 8*half + 2*(g&3) (+0/+1)
#pragma unroll
    for (int g = 0; g < 8; ++g) {
      int wo = ((g & 4) ? 8 : 0) + 4 * half + (g & 3);
      a.u[g] = AsU[m * 67 + k0h + wo];
    }
    // B 32x16 bf16 layout: VGPR j <-> K = (j>=4?16:0) + 8*half + 2*(j&3) (+0/+1), N = lane&15
#pragma unroll
    for (int j = 0; j < 8; ++j) {
      b.u[j] = Wp[(size_t)(k0h + ((j & 4) ? 8 : 0) + 4 * half + (j & 3)) * HD + n0 + m];
    }
    acc = __builtin_amdgcn_wmma_f32_16x16x32_bf16(false, a.v, false, b.v,
                                                  (short)0, acc, false, false);
  }

  const int nc = n0 + m;
  const float bv = bias ? bias[nc] : 0.0f;
#pragma unroll
  for (int r = 0; r < 8; ++r) {            // C/D: VGPR r <-> M = r + 8*half
    float v = acc[r] + bv;
    if (do_relu) v = fmaxf(v, 0.0f);
    out[(size_t)(row0 + r + 8 * half) * HD + nc] = v;
  }
}

// ---------- edge scatter: agg[dst] += xw[src] * norm  (one wave per edge) ----------
__global__ void edge_scatter(const float* __restrict__ xw, const int* __restrict__ src,
                             const int* __restrict__ dst, const float* __restrict__ nrm,
                             float* __restrict__ agg, int E) {
  int wave = (blockIdx.x * blockDim.x + threadIdx.x) >> 5;
  if (wave >= E) return;
  int lane = threadIdx.x & 31;
  int s = src[wave], d = dst[wave];
  float nm = nrm[wave];
  const float4 v = ((const float4*)(xw + (size_t)s * HD))[lane];
  float* ap = agg + (size_t)d * HD + lane * 4;
  atomicAdd(ap + 0, v.x * nm);
  atomicAdd(ap + 1, v.y * nm);
  atomicAdd(ap + 2, v.z * nm);
  atomicAdd(ap + 3, v.w * nm);
}

// ---------- h = relu(h + agg + xw*dinv^2 + bg)  (self-loop folded in, no atomics) ----------
__global__ void layer_update(float* __restrict__ h, const float* __restrict__ agg,
                             const float* __restrict__ xw, const float* __restrict__ dinv,
                             const float* __restrict__ bg, int n) {
  size_t idx = (size_t)blockIdx.x * blockDim.x + threadIdx.x;
  if (idx >= (size_t)n * HD) return;
  int i = (int)(idx >> 7);
  int c = (int)(idx & 127);
  float di = dinv[i];
  float v = h[idx] + agg[idx] + xw[idx] * di * di + bg[c];
  h[idx] = fmaxf(v, 0.0f);
}

// ---------- global add pool (one wave per node) ----------
__global__ void pool_add(const float* __restrict__ h, const int* __restrict__ batch,
                         float* __restrict__ pooled, int n) {
  int wave = (blockIdx.x * blockDim.x + threadIdx.x) >> 5;
  if (wave >= n) return;
  int lane = threadIdx.x & 31;
  int g = batch[wave];
  const float4 v = ((const float4*)(h + (size_t)wave * HD))[lane];
  float* pp = pooled + (size_t)g * HD + lane * 4;
  atomicAdd(pp + 0, v.x);
  atomicAdd(pp + 1, v.y);
  atomicAdd(pp + 2, v.z);
  atomicAdd(pp + 3, v.w);
}

// ---------- z @ Wc2 + bc2, softmax over 2 classes ----------
__global__ void classify_softmax(const float* __restrict__ z, const float* __restrict__ Wc2,
                                 const float* __restrict__ bc2, float* __restrict__ out, int G) {
  int g = blockIdx.x * blockDim.x + threadIdx.x;
  if (g >= G) return;
  float s0 = bc2[0], s1 = bc2[1];
#pragma unroll 8
  for (int k = 0; k < HD; ++k) {
    float v = z[g * HD + k];
    s0 += v * Wc2[k * 2 + 0];
    s1 += v * Wc2[k * 2 + 1];
  }
  float mx = fmaxf(s0, s1);
  float e0 = __expf(s0 - mx), e1 = __expf(s1 - mx);
  float inv = 1.0f / (e0 + e1);
  out[g * 2 + 0] = e0 * inv;
  out[g * 2 + 1] = e1 * inv;
}

// ---------- host ----------
extern "C" void kernel_launch(void* const* d_in, const int* in_sizes, int n_in,
                              void* d_out, int out_size, void* d_ws, size_t ws_size,
                              hipStream_t stream) {
  (void)n_in; (void)ws_size;
  const float* x     = (const float*)d_in[0];
  const int*   ei    = (const int*)  d_in[1];   // [2, E]: src then dst
  const int*   batch = (const int*)  d_in[2];
  const float* W0    = (const float*)d_in[3];
  const float* b0    = (const float*)d_in[4];
  const float* Wg    = (const float*)d_in[5];   // [3,128,128]
  const float* bg    = (const float*)d_in[6];   // [3,128]
  const float* Wc1   = (const float*)d_in[7];
  const float* bc1   = (const float*)d_in[8];
  const float* Wc2   = (const float*)d_in[9];
  const float* bc2   = (const float*)d_in[10];

  const int Nn = in_sizes[0] / HD;     // 100000 (multiple of 16)
  const int E  = in_sizes[1] / 2;      // 1600000
  const int G  = out_size / 2;         // 64 (multiple of 16)

  const int* src = ei;
  const int* dst = ei + E;

  // --- carve workspace ---
  char* w = (char*)d_ws;
  auto carve = [&](size_t bytes) -> void* {
    void* p = (void*)w;
    w += (bytes + 255) & ~(size_t)255;
    return p;
  };
  float*    dinv   = (float*)   carve((size_t)Nn * 4);
  float*    nrm    = (float*)   carve((size_t)E * 4);
  float*    h      = (float*)   carve((size_t)Nn * HD * 4);
  float*    xw     = (float*)   carve((size_t)Nn * HD * 4);
  float*    agg    = (float*)   carve((size_t)Nn * HD * 4);
  unsigned* Wpack  = (unsigned*)carve((size_t)(2 + NLAYERS) * 64 * HD * 4); // W0, Wg[0..2], Wc1
  float*    pooled = (float*)   carve((size_t)G * HD * 4);
  float*    z      = (float*)   carve((size_t)G * HD * 4);

  const int PW = 64 * HD;              // dwords per packed weight
  const int t256 = 256;

  // --- pack weights to bf16 pairs ---
  pack_weight_bf16<<<(PW + t256 - 1) / t256, t256, 0, stream>>>(W0,  Wpack + 0 * PW, PW);
  for (int l = 0; l < NLAYERS; ++l)
    pack_weight_bf16<<<(PW + t256 - 1) / t256, t256, 0, stream>>>(Wg + (size_t)l * HD * HD,
                                                                  Wpack + (1 + l) * PW, PW);
  pack_weight_bf16<<<(PW + t256 - 1) / t256, t256, 0, stream>>>(Wc1, Wpack + (1 + NLAYERS) * PW, PW);

  // --- gcn_norm: deg (with self-loops) -> dinv -> per-edge norm ---
  deg_init  <<<(Nn + t256 - 1) / t256, t256, 0, stream>>>(dinv, Nn);
  deg_accum <<<(E  + t256 - 1) / t256, t256, 0, stream>>>(dst, dinv, E);
  deg_rsqrt <<<(Nn + t256 - 1) / t256, t256, 0, stream>>>(dinv, Nn);
  edge_norm <<<(E  + t256 - 1) / t256, t256, 0, stream>>>(src, dst, dinv, nrm, E);

  // --- input linear + relu: h = relu(x @ W0 + b0) ---
  gemm_bf16_wmma<<<Nn / 16, 256, 0, stream>>>(x, Wpack + 0 * PW, b0, h, 1);

  // --- 3 GCN layers ---
  const int scatterBlocks = (E + 7) / 8;     // 8 waves (edges) per 256-thread block
  for (int l = 0; l < NLAYERS; ++l) {
    gemm_bf16_wmma<<<Nn / 16, 256, 0, stream>>>(h, Wpack + (1 + l) * PW, nullptr, xw, 0);
    hipMemsetAsync(agg, 0, (size_t)Nn * HD * 4, stream);
    edge_scatter<<<scatterBlocks, 256, 0, stream>>>(xw, src, dst, nrm, agg, E);
    layer_update<<<((size_t)Nn * HD + t256 - 1) / t256, t256, 0, stream>>>(
        h, agg, xw, dinv, bg + (size_t)l * HD, Nn);
  }

  // --- global add pool ---
  hipMemsetAsync(pooled, 0, (size_t)G * HD * 4, stream);
  pool_add<<<(Nn + 7) / 8, 256, 0, stream>>>(h, batch, pooled, Nn);

  // --- classifier: z = relu(pooled @ Wc1 + bc1); softmax(z @ Wc2 + bc2) ---
  gemm_bf16_wmma<<<G / 16, 256, 0, stream>>>(pooled, Wpack + (1 + NLAYERS) * PW, bc1, z, 1);
  classify_softmax<<<1, 64, 0, stream>>>(z, Wc2, bc2, (float*)d_out, G);
}
